// ResidualBlock_55448027791788
// MI455X (gfx1250) — matrix operand order repaired and checked by reading, not compiled, not run on hardware
//
#include <hip/hip_runtime.h>

// ---------------- problem constants (match reference) ----------------
#define D_MODEL 1024
#define D_INNER 2048
#define D_STATE 16
#define DT_RANK 64
#define D_CONV  4
#define SEQ_L   2048
#define XPROJ_N   (DT_RANK + 2 * D_STATE)   // 96 (logical)
#define XPROJ_PAD 128                       // padded for guard-free GEMM

typedef __attribute__((ext_vector_type(16))) __bf16 v16bf;
typedef __attribute__((ext_vector_type(8)))  __bf16 v8bf;
typedef __attribute__((ext_vector_type(8)))  float  v8f;

static __device__ __forceinline__ float sigmoidf_(float x) {
    return 1.0f / (1.0f + __expf(-x));
}

// ---------------- f32 -> bf16 convert ----------------
__global__ void cvt_f32_bf16(const float* __restrict__ in, __bf16* __restrict__ out, int n) {
    int i = blockIdx.x * blockDim.x + threadIdx.x;
    if (i < n) out[i] = (__bf16)in[i];
}

// x_proj_w [D_INNER, 96] -> bf16 [D_INNER, 128], zero-padded cols 96..127
__global__ void cvt_pad_xproj(const float* __restrict__ in, __bf16* __restrict__ out) {
    int i = blockIdx.x * blockDim.x + threadIdx.x;
    if (i >= D_INNER * XPROJ_PAD) return;
    int r = i >> 7, c = i & (XPROJ_PAD - 1);
    out[i] = (c < XPROJ_N) ? (__bf16)in[r * XPROJ_N + c] : (__bf16)0.0f;
}

// ---------------- RMSNorm -> bf16 ----------------
__global__ void rmsnorm_kernel(const float* __restrict__ x, const float* __restrict__ w,
                               __bf16* __restrict__ h) {
    __shared__ float red[8];
    int row = blockIdx.x;
    const float* xr = x + (size_t)row * D_MODEL;
    float ss = 0.f;
    for (int i = threadIdx.x; i < D_MODEL; i += blockDim.x) { float v = xr[i]; ss += v * v; }
    for (int off = 16; off; off >>= 1) ss += __shfl_xor(ss, off, 32);
    int wave = threadIdx.x >> 5, lane = threadIdx.x & 31;
    if (lane == 0) red[wave] = ss;
    __syncthreads();
    if (wave == 0) {
        float v = (lane < 8) ? red[lane] : 0.f;
        for (int off = 4; off; off >>= 1) v += __shfl_xor(v, off, 32);
        if (lane == 0) red[0] = v;
    }
    __syncthreads();
    float scale = rsqrtf(red[0] / (float)D_MODEL + 1e-5f);
    for (int i = threadIdx.x; i < D_MODEL; i += blockDim.x)
        h[(size_t)row * D_MODEL + i] = (__bf16)(xr[i] * scale * w[i]);
}

// ---------------- guard-free bf16 WMMA GEMM ----------------
// C[M,N](f32) = A[M,K](bf16,rm) @ B[K,N](bf16,rm).  Requires M%128==0, N%128==0, K%32==0.
// mode 0: plain   mode 1: softplus(acc + bias[col])   mode 2: acc + resid[row*N+col]
//
// Block tile 128x128 (8 waves), wave tile 64x32: 4 A-frags x 2 B-frags = 8 WMMA / K-step.
// LDS holds tiles pre-permuted into the WMMA per-lane layout (ISA 7.12.2):
//   A frag (16x32, 16-bit): lane = m + 16*h; chunk0 = K[8h..8h+8), chunk1 = K[16+8h..+8)
//   B frag (32x16, 16-bit): lane = n + 16*h; chunk0 = K[8h..8h+8), chunk1 = K[16+8h..+8)
// Layout [tile][chunk][lane][8]: per-lane stride 16B -> lanes 0..15 cover all 64 banks.
__global__ __launch_bounds__(256) void gemm_bf16_wmma(
    const __bf16* __restrict__ A, const __bf16* __restrict__ B,
    float* __restrict__ C, int M, int N, int K,
    int mode, const float* __restrict__ bias, const float* __restrict__ resid)
{
    __shared__ __align__(16) __bf16 AsP[8][2][32][8];   // 8 M-tiles of 16x32 (8 KB)
    __shared__ __align__(16) __bf16 BsP[8][2][32][8];   // 8 N-tiles of 32x16 (8 KB)

    const int tid  = threadIdx.x;
    const int lane = tid & 31, wave = tid >> 5;
    const int wm = wave >> 2, wn = wave & 3;            // 2 x 4 waves over 128x128
    const int half = lane >> 4, l16 = lane & 15;
    const int blockM = blockIdx.y * 128, blockN = blockIdx.x * 128;

    // A staging: thread -> rows (ar, ar+64), k-chunk kk -> one 16B LDS chunk each
    const int ar = tid >> 2, kk = (tid & 3) << 3;
    const int a_tile = ar >> 4, a_lane = (ar & 15) + ((kk & 8) << 1);  // m + 16*half
    const int a_chunk = kk >> 4;
    // B staging: thread -> k-row br, col chunks (bc, bc+64) -> scattered b16 (transpose)
    const int br = tid >> 3, bc = (tid & 7) << 3;
    const int b_tile0 = bc >> 4, b_n0 = (bc & 15);
    const int b_chunk = br >> 4, b_e7 = br & 7, b_h16 = ((br >> 3) & 1) << 4;

    v8f acc[4][2] = {};

    const __bf16* Ag0 = A + (size_t)(blockM + ar) * K + kk;
    const __bf16* Ag1 = Ag0 + (size_t)64 * K;
    const __bf16* Bg0 = B + (size_t)br * N + blockN + bc;
    const __bf16* Bg1 = Bg0 + 64;

    for (int k0 = 0; k0 < K; k0 += 32) {
        v8bf av0 = *(const v8bf*)(Ag0 + k0);
        v8bf av1 = *(const v8bf*)(Ag1 + k0);
        v8bf bv0 = *(const v8bf*)(Bg0 + (size_t)k0 * N);
        v8bf bv1 = *(const v8bf*)(Bg1 + (size_t)k0 * N);
        if (k0 + 32 < K) {
            __builtin_prefetch(Ag0 + k0 + 32, 0, 3);
            __builtin_prefetch(Ag1 + k0 + 32, 0, 3);
            __builtin_prefetch(Bg0 + (size_t)(k0 + 32) * N, 0, 3);
            __builtin_prefetch(Bg1 + (size_t)(k0 + 32) * N, 0, 3);
        }
        *(v8bf*)&AsP[a_tile    ][a_chunk][a_lane][0] = av0;
        *(v8bf*)&AsP[a_tile + 4][a_chunk][a_lane][0] = av1;
        #pragma unroll
        for (int t = 0; t < 8; ++t) {
            BsP[b_tile0    ][b_chunk][b_n0 + t + b_h16][b_e7] = bv0[t];
            BsP[b_tile0 + 4][b_chunk][b_n0 + t + b_h16][b_e7] = bv1[t];
        }
        __syncthreads();

        v16bf af[4], bfr[2];
        #pragma unroll
        for (int mi = 0; mi < 4; ++mi) {
            v8bf lo = *(const v8bf*)&AsP[4 * wm + mi][0][lane][0];
            v8bf hi = *(const v8bf*)&AsP[4 * wm + mi][1][lane][0];
            af[mi] = __builtin_shufflevector(lo, hi, 0,1,2,3,4,5,6,7,8,9,10,11,12,13,14,15);
        }
        #pragma unroll
        for (int ni = 0; ni < 2; ++ni) {
            v8bf lo = *(const v8bf*)&BsP[2 * wn + ni][0][lane][0];
            v8bf hi = *(const v8bf*)&BsP[2 * wn + ni][1][lane][0];
            bfr[ni] = __builtin_shufflevector(lo, hi, 0,1,2,3,4,5,6,7,8,9,10,11,12,13,14,15);
        }
        #pragma unroll
        for (int mi = 0; mi < 4; ++mi)
            #pragma unroll
            for (int ni = 0; ni < 2; ++ni)
                acc[mi][ni] = __builtin_amdgcn_wmma_f32_16x16x32_bf16(
                    false, af[mi], false, bfr[ni], (short)0, acc[mi][ni], false, false);
        __syncthreads();
    }

    // C/D layout: VGPR r -> lanes 0-15: M=r, N=lane; lanes 16-31: M=8+r
    #pragma unroll
    for (int ni = 0; ni < 2; ++ni) {
        const int col = blockN + wn * 32 + ni * 16 + l16;
        const float bcol = (mode == 1) ? bias[col] : 0.f;
        #pragma unroll
        for (int mi = 0; mi < 4; ++mi) {
            #pragma unroll
            for (int r = 0; r < 8; ++r) {
                int row = blockM + wm * 64 + mi * 16 + (half << 3) + r;
                float v = acc[mi][ni][r];
                if (mode == 1) {
                    v += bcol;
                    v = (v > 20.f) ? v : log1pf(__expf(v));
                } else if (mode == 2) {
                    v += resid[(size_t)row * N + col];
                }
                C[(size_t)row * N + col] = v;
            }
        }
    }
}

// ---------------- causal depthwise conv + SiLU ----------------
__global__ void conv_silu_kernel(const float* __restrict__ xr, const float* __restrict__ cw,
                                 const float* __restrict__ cb,
                                 float* __restrict__ uf, __bf16* __restrict__ ub) {
    int idx = blockIdx.x * blockDim.x + threadIdx.x;
    if (idx >= SEQ_L * D_INNER) return;
    int l = idx / D_INNER, d = idx - l * D_INNER;
    float acc = cb[d];
    #pragma unroll
    for (int t = 0; t < D_CONV; ++t) {
        int ll = l + t - (D_CONV - 1);
        if (ll >= 0) acc += xr[(size_t)ll * (2 * D_INNER) + d] * cw[d * D_CONV + t];
    }
    float s = acc * sigmoidf_(acc);
    uf[idx] = s;
    ub[idx] = (__bf16)s;
}

// ---------------- extract dt columns of x_dbl -> bf16 ----------------
__global__ void extract_dt(const float* __restrict__ xdbl, __bf16* __restrict__ dtb) {
    int idx = blockIdx.x * blockDim.x + threadIdx.x;
    if (idx >= SEQ_L * DT_RANK) return;
    int l = idx / DT_RANK, j = idx - l * DT_RANK;
    dtb[idx] = (__bf16)xdbl[(size_t)l * XPROJ_PAD + j];
}

// ---------------- selective scan ----------------
// one thread per (channel d, state n); wave32 = two channels; 16-lane xor reduce for y
__global__ void scan_kernel(const float* __restrict__ delta, const float* __restrict__ uf,
                            const float* __restrict__ xdbl, const float* __restrict__ A_log,
                            const float* __restrict__ Dv, float* __restrict__ y) {
    int gid = blockIdx.x * blockDim.x + threadIdx.x;
    int d = gid >> 4, n = gid & 15;
    if (d >= D_INNER) return;
    float An = -__expf(A_log[d * D_STATE + n]);
    float Dd = Dv[d];
    float s = 0.f;
    for (int l = 0; l < SEQ_L; ++l) {
        const float* xrow = xdbl + (size_t)l * XPROJ_PAD;
        float dl = delta[(size_t)l * D_INNER + d];
        float ul = uf[(size_t)l * D_INNER + d];
        float b  = xrow[DT_RANK + n];
        float c  = xrow[DT_RANK + D_STATE + n];
        s = __expf(dl * An) * s + (dl * ul) * b;
        float part = s * c;
        part += __shfl_xor(part, 1, 32);
        part += __shfl_xor(part, 2, 32);
        part += __shfl_xor(part, 4, 32);
        part += __shfl_xor(part, 8, 32);
        if (n == 0) y[(size_t)l * D_INNER + d] = part + ul * Dd;
    }
}

// ---------------- gate: y * silu(res) -> bf16 ----------------
__global__ void gate_kernel(const float* __restrict__ xr, const float* __restrict__ y,
                            __bf16* __restrict__ yg) {
    int idx = blockIdx.x * blockDim.x + threadIdx.x;
    if (idx >= SEQ_L * D_INNER) return;
    int l = idx / D_INNER, d = idx - l * D_INNER;
    float res = xr[(size_t)l * (2 * D_INNER) + D_INNER + d];
    yg[idx] = (__bf16)(y[idx] * res * sigmoidf_(res));
}

// ---------------- host launch ----------------
extern "C" void kernel_launch(void* const* d_in, const int* in_sizes, int n_in,
                              void* d_out, int out_size, void* d_ws, size_t ws_size,
                              hipStream_t stream) {
    const float* x       = (const float*)d_in[0];   // [L, D_MODEL]
    const float* in_w    = (const float*)d_in[1];   // [D_MODEL, 2*D_INNER]
    const float* conv_w  = (const float*)d_in[2];   // [D_INNER, 1, D_CONV]
    const float* conv_b  = (const float*)d_in[3];   // [D_INNER]
    const float* xproj_w = (const float*)d_in[4];   // [D_INNER, 96]
    const float* dt_w    = (const float*)d_in[5];   // [DT_RANK, D_INNER]
    const float* dt_bias = (const float*)d_in[6];   // [D_INNER]
    const float* A_log   = (const float*)d_in[7];   // [D_INNER, D_STATE]
    const float* Dv      = (const float*)d_in[8];   // [D_INNER]
    const float* out_w   = (const float*)d_in[9];   // [D_INNER, D_MODEL]
    const float* norm_w  = (const float*)d_in[10];  // [D_MODEL]
    float* out = (float*)d_out;

    char* ws = (char*)d_ws;
    size_t off = 0;
    auto carve = [&](size_t bytes) -> char* {
        char* p = ws + off;
        off = (off + bytes + 255) & ~(size_t)255;
        return p;
    };
    __bf16* h_b     = (__bf16*)carve((size_t)SEQ_L * D_MODEL * 2);
    __bf16* in_w_b  = (__bf16*)carve((size_t)D_MODEL * 2 * D_INNER * 2);
    __bf16* xpw_b   = (__bf16*)carve((size_t)D_INNER * XPROJ_PAD * 2);
    __bf16* dtw_b   = (__bf16*)carve((size_t)DT_RANK * D_INNER * 2);
    __bf16* outw_b  = (__bf16*)carve((size_t)D_INNER * D_MODEL * 2);
    float*  xr      = (float*) carve((size_t)SEQ_L * 2 * D_INNER * 4);
    float*  u_f     = (float*) carve((size_t)SEQ_L * D_INNER * 4);
    __bf16* u_b     = (__bf16*)carve((size_t)SEQ_L * D_INNER * 2);
    float*  xdbl    = (float*) carve((size_t)SEQ_L * XPROJ_PAD * 4);
    __bf16* dt_b16  = (__bf16*)carve((size_t)SEQ_L * DT_RANK * 2);
    float*  delta   = (float*) carve((size_t)SEQ_L * D_INNER * 4);
    float*  yv      = (float*) carve((size_t)SEQ_L * D_INNER * 4);
    __bf16* yg_b    = (__bf16*)carve((size_t)SEQ_L * D_INNER * 2);
    (void)ws_size; (void)n_in; (void)in_sizes; (void)out_size;

    const int T = 256;
    auto blocks = [](int n, int t) { return (n + t - 1) / t; };

    // 1) weight conversions to bf16 (x_proj padded 96 -> 128)
    cvt_f32_bf16<<<blocks(D_MODEL * 2 * D_INNER, T), T, 0, stream>>>(in_w, in_w_b, D_MODEL * 2 * D_INNER);
    cvt_pad_xproj<<<blocks(D_INNER * XPROJ_PAD, T), T, 0, stream>>>(xproj_w, xpw_b);
    cvt_f32_bf16<<<blocks(DT_RANK * D_INNER, T), T, 0, stream>>>(dt_w, dtw_b, DT_RANK * D_INNER);
    cvt_f32_bf16<<<blocks(D_INNER * D_MODEL, T), T, 0, stream>>>(out_w, outw_b, D_INNER * D_MODEL);

    // 2) RMSNorm -> h (bf16)
    rmsnorm_kernel<<<SEQ_L, T, 0, stream>>>(x, norm_w, h_b);

    // 3) in_proj: xr[L, 4096] = h @ in_w
    gemm_bf16_wmma<<<dim3((2 * D_INNER) / 128, SEQ_L / 128), T, 0, stream>>>(
        h_b, in_w_b, xr, SEQ_L, 2 * D_INNER, D_MODEL, 0, nullptr, nullptr);

    // 4) causal depthwise conv + SiLU
    conv_silu_kernel<<<blocks(SEQ_L * D_INNER, T), T, 0, stream>>>(xr, conv_w, conv_b, u_f, u_b);

    // 5) x_proj: xdbl[L, 128pad] = u @ x_proj_w_pad
    gemm_bf16_wmma<<<dim3(XPROJ_PAD / 128, SEQ_L / 128), T, 0, stream>>>(
        u_b, xpw_b, xdbl, SEQ_L, XPROJ_PAD, D_INNER, 0, nullptr, nullptr);

    // 6) dt slice -> bf16
    extract_dt<<<blocks(SEQ_L * DT_RANK, T), T, 0, stream>>>(xdbl, dt_b16);

    // 7) delta = softplus(dt @ dt_w + dt_b)
    gemm_bf16_wmma<<<dim3(D_INNER / 128, SEQ_L / 128), T, 0, stream>>>(
        dt_b16, dtw_b, delta, SEQ_L, D_INNER, DT_RANK, 1, dt_bias, nullptr);

    // 8) selective scan -> y
    scan_kernel<<<blocks(D_INNER * D_STATE, T), T, 0, stream>>>(delta, u_f, xdbl, A_log, Dv, yv);

    // 9) gate: yg = y * silu(res)
    gate_kernel<<<blocks(SEQ_L * D_INNER, T), T, 0, stream>>>(xr, yv, yg_b);

    // 10) out_proj + residual: out = yg @ out_w + x
    gemm_bf16_wmma<<<dim3(D_MODEL / 128, SEQ_L / 128), T, 0, stream>>>(
        yg_b, outw_b, out, SEQ_L, D_MODEL, D_INNER, 2, nullptr, x);
}
